// CPIFuse_36558761623727
// MI455X (gfx1250) — compile-verified, matrix operand order checked by reading, not run on hardware
//
#include <hip/hip_runtime.h>
#include <cstdint>
#include <cstddef>

// ---------------------------------------------------------------------------
// CPIFuse for MI455X (gfx1250): bf16 WMMA GEMMs (16x16x32, f32 accum) for all
// 1x1 convs, K-split WMMA Gram for channel attention, fused depthwise+GELU
// gate for the FFN. Weights pre-converted to padded bf16 once, then staged
// into LDS per-workgroup via the Tensor Data Mover (tensor_load_to_lds +
// s_wait_tensorcnt). Intermediates in bf16 (bandwidth-bound at 23.3 TB/s).
// ---------------------------------------------------------------------------

#define NSP   65536      // pixels per batch (256*256)
#define NPIX  131072     // total pixels (B=2)
#define CDIM  96
#define HEADS 8
#define CH    12         // channels per head
#define HID   255
#define HID2  510

typedef __attribute__((ext_vector_type(16))) __bf16 v16bf;
typedef __attribute__((ext_vector_type(8)))  float  v8f;
typedef __attribute__((ext_vector_type(4)))  unsigned int u32x4;
typedef __attribute__((ext_vector_type(8)))  int i32x8;
typedef __attribute__((ext_vector_type(4)))  int i32x4;

union Frag16 { v16bf v; uint4 u[2]; };

// ---------------------------------------------------------------------------
// Convert f32 weight [Nreal, Kreal] -> zero-padded bf16 [Npad, KT]
// ---------------------------------------------------------------------------
__global__ void wconv_kernel(const float* __restrict__ Wg, __bf16* __restrict__ Wb,
                             int Nreal, int Kreal, int Npad, int KT) {
  int i = blockIdx.x * blockDim.x + threadIdx.x;
  int total = Npad * KT;
  if (i >= total) return;
  int r = i / KT, k = i - r * KT;
  if (i + 256 < total) __builtin_prefetch(&Wg[(size_t)r * Kreal + k], 0, 1);
  float w = (r < Nreal && k < Kreal) ? Wg[(size_t)r * Kreal + k] : 0.f;
  Wb[i] = (__bf16)w;
}

// ---------------------------------------------------------------------------
// LayerNorm over channels of NCHW f32 -> pixel-major bf16 [NP, 96]
// ---------------------------------------------------------------------------
__global__ void ln_kernel(const float* __restrict__ in, const float* __restrict__ w,
                          const float* __restrict__ bias, __bf16* __restrict__ out) {
  int p = blockIdx.x * blockDim.x + threadIdx.x;       // 0..NPIX-1
  int b = p >> 16, n = p & 65535;
  const float* src = in + (((size_t)b * CDIM) << 16) + n;
  float v[CDIM]; float s = 0.f, s2 = 0.f;
#pragma unroll
  for (int c = 0; c < CDIM; ++c) { float t = src[(size_t)c << 16]; v[c] = t; s += t; s2 += t * t; }
  float mu  = s * (1.f / CDIM);
  float var = s2 * (1.f / CDIM) - mu * mu;
  float rs  = rsqrtf(var + 1e-5f);
  __bf16* dst = out + (size_t)p * CDIM;
#pragma unroll
  for (int c = 0; c < CDIM; ++c) dst[c] = (__bf16)((v[c] - mu) * rs * w[c] + bias[c]);
}

// ---------------------------------------------------------------------------
// Generic pixel-batched GEMM: Y[M,N] = X[M,K] * W[N,K]^T, bf16 in, f32 acc.
//   MODE 1: store bf16 row-major (ldy, colofs)
//   MODE 2: add bf16 residual [m*96+n], store bf16 row-major (ldy, colofs)
//   MODE 3: add f32 NCHW residual, store f32 NCHW
// Pre-padded bf16 weights [Npad, KT] DMA'd to LDS by the Tensor Data Mover
// (wave 0 issues the descriptor; TENSORcnt wait; workgroup barrier).
// Block = 256 threads (8 waves); each wave owns a 16-row M tile; A fragments
// stay resident in VGPRs across the N loop.
// ---------------------------------------------------------------------------
template <int KT, int MODE>
__global__ void gemm_ws(const __bf16* __restrict__ X, int ldx,
                        const __bf16* __restrict__ Wbf, int Npad,
                        __bf16* __restrict__ outb, float* __restrict__ outf,
                        const __bf16* __restrict__ resb, const float* __restrict__ resf,
                        int ldy, int colofs) {
  extern __shared__ __bf16 wl[];                       // Npad * KT bf16
  const int tid = threadIdx.x;
  const int nelem = Npad * KT;                         // <= 49152 (< 2^16)

  if (tid < 32) {
    // --- TDM: flat 1-D tile copy global bf16 -> LDS (D# groups per ISA §8) ---
    uint64_t ga = (uint64_t)(uintptr_t)Wbf;
    u32x4 g0;
    g0[0] = 1u;                                        // count=1, user desc
    g0[1] = 0u;                                        // lds_addr = 0 (bytes)
    g0[2] = (unsigned int)(ga & 0xFFFFFFFFu);          // global_addr[31:0]
    g0[3] = (unsigned int)((ga >> 32) & 0x1FFFFFFu) | (2u << 30);  // addr hi | type=2
    i32x8 g1;
    g1[0] = (int)(1u << 16);                           // data_size=1 (2 bytes)
    g1[1] = (int)(((unsigned)nelem & 0xFFFFu) << 16);  // tensor_dim0[15:0]
    g1[2] = (int)(((unsigned)nelem >> 16) & 0xFFFFu);  // tensor_dim0[31:16]
    g1[3] = (int)(((unsigned)nelem & 0xFFFFu) << 16);  // tile_dim0
    g1[4] = 0;                                         // tile_dim1=0 (1-D), tile_dim2=0
    g1[5] = (int)nelem;                                // tensor_dim0_stride lo
    g1[6] = 0; g1[7] = 0;
    i32x4 g2 = {0, 0, 0, 0};
    i32x4 g3 = {0, 0, 0, 0};
    i32x8 g4 = {0, 0, 0, 0, 0, 0, 0, 0};
    __builtin_amdgcn_tensor_load_to_lds(g0, g1, g2, g3, g4, 0);
    __builtin_amdgcn_s_wait_tensorcnt(0);
  }
  __syncthreads();

  const int lane = tid & 31, wv = tid >> 5;
  const int m0   = blockIdx.x * 128 + wv * 16;
  const int mrow = m0 + (lane & 15);
  const int half = lane >> 4;
  constexpr int KTILES = KT / 32;

  Frag16 a[KTILES];
  const __bf16* xr = X + (size_t)mrow * ldx;
#pragma unroll
  for (int t = 0; t < KTILES; ++t) {
    a[t].u[0] = *reinterpret_cast<const uint4*>(xr + t * 32 + 8 * half);
    a[t].u[1] = *reinterpret_cast<const uint4*>(xr + t * 32 + 16 + 8 * half);
  }

  for (int n0 = 0; n0 < Npad; n0 += 16) {
    const __bf16* wr = wl + (size_t)(n0 + (lane & 15)) * KT;
    v8f acc = {};
#pragma unroll
    for (int t = 0; t < KTILES; ++t) {
      Frag16 bf;
      bf.u[0] = *reinterpret_cast<const uint4*>(wr + t * 32 + 8 * half);
      bf.u[1] = *reinterpret_cast<const uint4*>(wr + t * 32 + 16 + 8 * half);
      acc = __builtin_amdgcn_wmma_f32_16x16x32_bf16(false, a[t].v, false, bf.v,
                                                    (short)0, acc, false, false);
    }
    const int n = n0 + (lane & 15);
#pragma unroll
    for (int r = 0; r < 8; ++r) {
      const int m = m0 + r + 8 * half;
      float val = acc[r];
      if constexpr (MODE == 1) {
        outb[(size_t)m * ldy + colofs + n] = (__bf16)val;
      } else if constexpr (MODE == 2) {
        val += (float)resb[(size_t)m * CDIM + n];
        outb[(size_t)m * ldy + colofs + n] = (__bf16)val;
      } else if constexpr (MODE == 3) {
        int b = m >> 16, nn = m & 65535;
        size_t idx = (((size_t)(b * CDIM + n)) << 16) + nn;
        outf[idx] = val + resf[idx];
      }
    }
  }
}

// ---------------------------------------------------------------------------
// Depthwise 3x3 (SAME, zero pad) on pixel-major bf16 [NP, 288];
// writes channel-major bf16 [2, 288, 65536] (ready for Gram / attn@V).
// ---------------------------------------------------------------------------
__global__ void dw3_kernel(const __bf16* __restrict__ in, const float* __restrict__ w,
                           __bf16* __restrict__ outT) {
  const int n = blockIdx.x * blockDim.x + threadIdx.x;   // 0..65535
  const int c = blockIdx.y;                              // 0..287
  const int b = blockIdx.z;
  const int hy = n >> 8, wx = n & 255;
  float wt[9];
#pragma unroll
  for (int i = 0; i < 9; ++i) wt[i] = w[c * 9 + i];
  float s = 0.f;
#pragma unroll
  for (int dy = -1; dy <= 1; ++dy) {
#pragma unroll
    for (int dx = -1; dx <= 1; ++dx) {
      int yy = hy + dy, xx = wx + dx;
      if (yy >= 0 && yy < 256 && xx >= 0 && xx < 256) {
        int p = (b << 16) | (yy << 8) | xx;
        s += (float)in[(size_t)p * 288 + c] * wt[(dy + 1) * 3 + (dx + 1)];
      }
    }
  }
  outT[(((size_t)(b * 288 + c)) << 16) + n] = (__bf16)s;
}

// ---------------------------------------------------------------------------
// L2 norms over spatial dim for q/k rows: norms[branch][b][ch(0..191)]
// ---------------------------------------------------------------------------
__global__ void norms_kernel(const __bf16* __restrict__ qkvTA,
                             const __bf16* __restrict__ qkvTB,
                             float* __restrict__ norms) {
  __shared__ float red[256];
  const int idx = blockIdx.x;                 // 0..767
  const int branch = idx / 384;
  const int rem = idx % 384;
  const int b = rem / 192, ch = rem % 192;    // ch<96: q row, >=96: k row
  const __bf16* src = (branch ? qkvTB : qkvTA) + (((size_t)(b * 288 + ch)) << 16);
  float s = 0.f;
  for (int i = threadIdx.x; i < NSP; i += 256) { float t = (float)src[i]; s += t * t; }
  red[threadIdx.x] = s;
  __syncthreads();
  for (int st = 128; st > 0; st >>= 1) {
    if (threadIdx.x < st) red[threadIdx.x] += red[threadIdx.x + st];
    __syncthreads();
  }
  if (threadIdx.x == 0) norms[idx] = fmaxf(sqrtf(red[0]), 1e-12f);
}

__global__ void zero_kernel(float* __restrict__ p, int n) {
  int i = blockIdx.x * blockDim.x + threadIdx.x;
  if (i < n) p[i] = 0.f;
}

// ---------------------------------------------------------------------------
// WMMA Gram: S[c,d] = sum_n Q[c,n] * K[d,n]  (per pair/b/head), K-split with
// f32 global atomics. One wave per block; chunk = 1024 spatial elements.
//   pair 0: Q = qkvT_B q-section, K = qkvT_A k-section  (attn_a)
//   pair 1: Q = qkvT_A q-section, K = qkvT_B k-section  (attn_b)
// ---------------------------------------------------------------------------
__global__ void gram_kernel(const __bf16* __restrict__ qkvTA,
                            const __bf16* __restrict__ qkvTB,
                            float* __restrict__ gram) {
  const int id = blockIdx.x;                // 0..31
  const int pair = id >> 4, b = (id >> 3) & 1, h = id & 7;
  const int chunk = blockIdx.y;             // 0..63
  const int lane = threadIdx.x & 31;
  const int half = lane >> 4;

  const __bf16* Qbase = (pair == 0) ? qkvTB : qkvTA;
  const __bf16* Kbase = (pair == 0) ? qkvTA : qkvTB;
  const int qrow = b * 288 + h * CH + (lane & 15);
  const int krow = b * 288 + 96 + h * CH + (lane & 15);
  const __bf16* qp = Qbase + (((size_t)qrow) << 16);
  const __bf16* kp = Kbase + (((size_t)krow) << 16);

  v8f acc = {};
  const int nbase = chunk * 1024;
#pragma unroll 4
  for (int nt = 0; nt < 32; ++nt) {
    const int n0 = nbase + nt * 32;
    Frag16 af, bf;
    af.u[0] = *reinterpret_cast<const uint4*>(qp + n0 + 8 * half);
    af.u[1] = *reinterpret_cast<const uint4*>(qp + n0 + 16 + 8 * half);
    bf.u[0] = *reinterpret_cast<const uint4*>(kp + n0 + 8 * half);
    bf.u[1] = *reinterpret_cast<const uint4*>(kp + n0 + 16 + 8 * half);
    acc = __builtin_amdgcn_wmma_f32_16x16x32_bf16(false, af.v, false, bf.v,
                                                  (short)0, acc, false, false);
  }
  float* g = gram + (size_t)id * 256;
#pragma unroll
  for (int r = 0; r < 8; ++r) {
    int m = r + 8 * half;                       // q-channel c
    atomicAdd(&g[m * 16 + (lane & 15)], acc[r]);
  }
}

// ---------------------------------------------------------------------------
// L2-normalize Gram + temp scale (+/- sign) + softmax over d.
// attn layout: [pair][b][head][12][12] f32
// ---------------------------------------------------------------------------
__global__ void softmax_kernel(const float* __restrict__ gram,
                               const float* __restrict__ norms,
                               const float* __restrict__ temp,
                               float* __restrict__ attn) {
  const int id = blockIdx.x;                // 0..31
  const int pair = id >> 4, b = (id >> 3) & 1, h = id & 7;
  const int c = threadIdx.x;
  if (c >= CH) return;
  const int qbr = (pair == 0) ? 1 : 0;      // branch of Q
  const int kbr = (pair == 0) ? 0 : 1;
  const float sign = (pair == 0) ? 1.f : -1.f;
  const float tmp = temp[h];
  const float nq = norms[(qbr * 2 + b) * 192 + h * CH + c];
  const float* g = gram + (size_t)id * 256 + c * 16;
  float v[CH], mx = -1e30f;
#pragma unroll
  for (int d = 0; d < CH; ++d) {
    float nk = norms[(kbr * 2 + b) * 192 + 96 + h * CH + d];
    float s = sign * g[d] / (nq * nk) * tmp;
    v[d] = s; mx = fmaxf(mx, s);
  }
  float sum = 0.f;
#pragma unroll
  for (int d = 0; d < CH; ++d) { v[d] = __expf(v[d] - mx); sum += v[d]; }
  float inv = 1.f / sum;
  float* a = attn + (size_t)id * 144 + c * CH;
#pragma unroll
  for (int d = 0; d < CH; ++d) a[d] = v[d] * inv;
}

// ---------------------------------------------------------------------------
// out[c,n] = sum_d attn[c,d] * v[d,n]; v channel-major, out pixel-major bf16.
// ---------------------------------------------------------------------------
__global__ void attnv_kernel(const float* __restrict__ attn,
                             const __bf16* __restrict__ qkvT,
                             __bf16* __restrict__ out, int pair) {
  const int n = blockIdx.x * blockDim.x + threadIdx.x;   // 0..65535
  const int c = blockIdx.y;                              // 0..95
  const int b = blockIdx.z;
  const int h = c / CH, cc = c % CH;
  const float* a = attn + (size_t)(((pair * 2 + b) * HEADS + h)) * 144 + cc * CH;
  float acc = 0.f;
#pragma unroll
  for (int d = 0; d < CH; ++d) {
    const __bf16* vp = qkvT + (((size_t)(b * 288 + 192 + h * CH + d)) << 16);
    acc += a[d] * (float)vp[n];
  }
  out[(size_t)((b << 16) | n) * CDIM + c] = (__bf16)acc;
}

// ---------------------------------------------------------------------------
// FFN depthwise 3x3 + exact GELU gate: h[p,c] = gelu(dw(t,c)) * dw(t,c+255)
// t pixel-major bf16 [NP, 512] (510 valid); h pixel-major bf16 [NP, 256].
// ---------------------------------------------------------------------------
__global__ void dwf_gate_kernel(const __bf16* __restrict__ t,
                                const float* __restrict__ w,
                                __bf16* __restrict__ hout) {
  const int p = blockIdx.x;
  const int c = threadIdx.x;                 // 0..255
  if (c == HID) { hout[(size_t)p * 256 + c] = (__bf16)0.f; return; }
  const int b = p >> 16, n = p & 65535;
  const int hy = n >> 8, wx = n & 255;
  float w1[9], w2[9];
#pragma unroll
  for (int i = 0; i < 9; ++i) { w1[i] = w[c * 9 + i]; w2[i] = w[(c + HID) * 9 + i]; }
  float o1 = 0.f, o2 = 0.f;
#pragma unroll
  for (int dy = -1; dy <= 1; ++dy) {
#pragma unroll
    for (int dx = -1; dx <= 1; ++dx) {
      int yy = hy + dy, xx = wx + dx;
      if (yy >= 0 && yy < 256 && xx >= 0 && xx < 256) {
        size_t pp = (size_t)((b << 16) | (yy << 8) | xx) * 512;
        int wi = (dy + 1) * 3 + (dx + 1);
        o1 += (float)t[pp + c] * w1[wi];
        o2 += (float)t[pp + c + HID] * w2[wi];
      }
    }
  }
  float g = 0.5f * o1 * (1.f + erff(o1 * 0.70710678f));
  hout[(size_t)p * 256 + c] = (__bf16)(g * o2);
}

// ---------------------------------------------------------------------------
extern "C" void kernel_launch(void* const* d_in, const int* in_sizes, int n_in,
                              void* d_out, int out_size, void* d_ws, size_t ws_size,
                              hipStream_t stream) {
  const float* x       = (const float*)d_in[0];
  const float* y       = (const float*)d_in[1];
  const float* ln1_w   = (const float*)d_in[2];
  const float* ln1_b   = (const float*)d_in[3];
  const float* ln2_w   = (const float*)d_in[4];
  const float* ln2_b   = (const float*)d_in[5];
  const float* qkvA_w  = (const float*)d_in[6];
  const float* dwA_w   = (const float*)d_in[7];
  const float* qkvB_w  = (const float*)d_in[8];
  const float* dwB_w   = (const float*)d_in[9];
  const float* projA_w = (const float*)d_in[10];
  const float* projB_w = (const float*)d_in[11];
  const float* concat_w= (const float*)d_in[12];
  const float* temp    = (const float*)d_in[13];
  const float* pin_w   = (const float*)d_in[14];
  const float* dwf_w   = (const float*)d_in[15];
  const float* pout_w  = (const float*)d_in[16];
  float* out = (float*)d_out;

  // ---- workspace layout (bf16 intermediates; big slots reused) ----
  char* base = (char*)d_ws;
  size_t off = 0;
  auto alloc = [&](size_t bytes) { void* p = base + off; off += (bytes + 255) & ~(size_t)255; return p; };
  __bf16* xn      = (__bf16*)alloc((size_t)NPIX * CDIM * 2);
  __bf16* yn      = (__bf16*)alloc((size_t)NPIX * CDIM * 2);
  __bf16* qkv_pm  = (__bf16*)alloc((size_t)NPIX * 288 * 2);      // reused as h_pm
  __bf16* qkvTA   = (__bf16*)alloc((size_t)2 * 288 * NSP * 2);   // reused as t_pm (part 1)
  __bf16* qkvTB   = (__bf16*)alloc((size_t)2 * 288 * NSP * 2);   // reused as t_pm (part 2)
  __bf16* attnOut = (__bf16*)alloc((size_t)NPIX * CDIM * 2);
  __bf16* cat     = (__bf16*)alloc((size_t)NPIX * 192 * 2);
  float*  x1      = (float*) alloc((size_t)NPIX * CDIM * 4);
  __bf16* ln2buf  = (__bf16*)alloc((size_t)NPIX * CDIM * 2);
  float*  gram    = (float*) alloc(32 * 256 * 4);
  float*  norms   = (float*) alloc(768 * 4);
  float*  attn    = (float*) alloc(32 * 144 * 4);
  // pre-padded bf16 weights for TDM staging
  __bf16* wqA   = (__bf16*)alloc(288 * 96 * 2);
  __bf16* wqB   = (__bf16*)alloc(288 * 96 * 2);
  __bf16* wpA   = (__bf16*)alloc(96 * 96 * 2);
  __bf16* wpB   = (__bf16*)alloc(96 * 96 * 2);
  __bf16* wcat  = (__bf16*)alloc(96 * 192 * 2);
  __bf16* wpin  = (__bf16*)alloc(512 * 96 * 2);
  __bf16* wpout = (__bf16*)alloc(96 * 256 * 2);
  __bf16* t_pm  = qkvTA;   // [NP,512] bf16 = 128 MB <= 144 MB (qkvTA+qkvTB)
  __bf16* h_pm  = qkv_pm;  // [NP,256] bf16 = 64 MB <= 72 MB

  const dim3 blk256(256);

  // 0) one-time weight conversion f32 -> padded bf16 (tiny)
  wconv_kernel<<<(288 * 96 + 255) / 256, blk256, 0, stream>>>(qkvA_w, wqA, 288, 96, 288, 96);
  wconv_kernel<<<(288 * 96 + 255) / 256, blk256, 0, stream>>>(qkvB_w, wqB, 288, 96, 288, 96);
  wconv_kernel<<<(96 * 96 + 255) / 256, blk256, 0, stream>>>(projA_w, wpA, 96, 96, 96, 96);
  wconv_kernel<<<(96 * 96 + 255) / 256, blk256, 0, stream>>>(projB_w, wpB, 96, 96, 96, 96);
  wconv_kernel<<<(96 * 192 + 255) / 256, blk256, 0, stream>>>(concat_w, wcat, 96, 192, 96, 192);
  wconv_kernel<<<(512 * 96 + 255) / 256, blk256, 0, stream>>>(pin_w, wpin, 510, 96, 512, 96);
  wconv_kernel<<<(96 * 256 + 255) / 256, blk256, 0, stream>>>(pout_w, wpout, 96, 255, 96, 256);

  // 1) LayerNorm x, y (norm1 on both, as in reference)
  ln_kernel<<<NPIX / 256, blk256, 0, stream>>>(x, ln1_w, ln1_b, xn);
  ln_kernel<<<NPIX / 256, blk256, 0, stream>>>(y, ln1_w, ln1_b, yn);

  // 2) qkv GEMMs + depthwise, per branch (qkv_pm buffer reused)
  gemm_ws<96, 1><<<NPIX / 128, blk256, 288 * 96 * 2, stream>>>(
      xn, 96, wqA, 288, qkv_pm, nullptr, nullptr, nullptr, 288, 0);
  dw3_kernel<<<dim3(256, 288, 2), blk256, 0, stream>>>(qkv_pm, dwA_w, qkvTA);
  gemm_ws<96, 1><<<NPIX / 128, blk256, 288 * 96 * 2, stream>>>(
      yn, 96, wqB, 288, qkv_pm, nullptr, nullptr, nullptr, 288, 0);
  dw3_kernel<<<dim3(256, 288, 2), blk256, 0, stream>>>(qkv_pm, dwB_w, qkvTB);

  // 3) channel attention: norms, WMMA Gram (K-split + atomics), softmax
  zero_kernel<<<(32 * 256 + 255) / 256, blk256, 0, stream>>>(gram, 32 * 256);
  norms_kernel<<<768, blk256, 0, stream>>>(qkvTA, qkvTB, norms);
  gram_kernel<<<dim3(32, 64), dim3(32), 0, stream>>>(qkvTA, qkvTB, gram);
  softmax_kernel<<<32, dim3(32), 0, stream>>>(gram, norms, temp, attn);

  // 4) attn @ V, then proj (+ normalized residual) into concat buffer
  attnv_kernel<<<dim3(256, 96, 2), blk256, 0, stream>>>(attn, qkvTA, attnOut, 0);
  gemm_ws<96, 2><<<NPIX / 128, blk256, 96 * 96 * 2, stream>>>(
      attnOut, 96, wpA, 96, cat, nullptr, xn, nullptr, 192, 0);
  attnv_kernel<<<dim3(256, 96, 2), blk256, 0, stream>>>(attn, qkvTB, attnOut, 1);
  gemm_ws<96, 2><<<NPIX / 128, blk256, 96 * 96 * 2, stream>>>(
      attnOut, 96, wpB, 96, cat, nullptr, yn, nullptr, 192, 96);

  // 5) concat fusion: x1 = x + concat_w @ [out_a; out_b]   (f32 NCHW)
  gemm_ws<192, 3><<<NPIX / 128, blk256, 96 * 192 * 2, stream>>>(
      cat, 192, wcat, 96, nullptr, x1, nullptr, x, 0, 0);

  // 6) gated FFN: LN2 -> pin (510, padded 512) -> dw3+GELU gate -> pout (+x1)
  ln_kernel<<<NPIX / 256, blk256, 0, stream>>>(x1, ln2_w, ln2_b, ln2buf);
  gemm_ws<96, 1><<<NPIX / 128, blk256, 512 * 96 * 2, stream>>>(
      ln2buf, 96, wpin, 512, t_pm, nullptr, nullptr, nullptr, 512, 0);
  dwf_gate_kernel<<<NPIX, blk256, 0, stream>>>(t_pm, dwf_w, h_pm);
  gemm_ws<256, 3><<<NPIX / 128, blk256, 96 * 256 * 2, stream>>>(
      h_pm, 256, wpout, 96, nullptr, out, nullptr, x1, 0, 0);
}
